// PocketConditionedDenoiser_79422535238050
// MI455X (gfx1250) — matrix operand order repaired and checked by reading, not compiled, not run on hardware
//
#include <hip/hip_runtime.h>
#include <hip/hip_bf16.h>
#include <math.h>

#define DEVFN __device__ __forceinline__

typedef _Float16 h8    __attribute__((ext_vector_type(8)));
typedef _Float16 v16h_t __attribute__((ext_vector_type(16)));
typedef float    v8f_t  __attribute__((ext_vector_type(8)));

constexpr int Bn = 32, Nl = 64, Ee = 192, NP = 384, DP = 128, H = 256, KA = 16, KB = 5, KC = 32, L = 4;
constexpr int NLIG = Bn * Nl;    // 2048 ligand nodes
constexpr int NE   = Bn * Ee;    // 6144 edges
constexpr int NPK  = Bn * NP;    // 12288 pocket points
constexpr int NCR  = NLIG * KC;  // 65536 cross rows
constexpr int KMSG = 800;        // 772 padded to 32
constexpr int KCRS = 544;        // 516 padded to 32
constexpr int G3H  = 3 * H;      // 768
constexpr int RW   = 288;        // rest-row width (256 feat + 4 geom + pad)

constexpr int EPI_F16_RELU = 0, EPI_F16 = 1, EPI_F32 = 2, EPI_SCATTER = 3, EPI_DIV32 = 4;

DEVFN v16h_t mk16(h8 lo, h8 hi) {
  return __builtin_shufflevector(lo, hi, 0,1,2,3,4,5,6,7,8,9,10,11,12,13,14,15);
}

// ---------------- A-matrix loaders (return 8 contiguous f16 at [row, k..k+7]) ----------
struct PlainAL {
  const _Float16* A; int ld;
  DEVFN h8 load8(int row, int k) const { return *(const h8*)(A + (size_t)row * ld + k); }
};
// msg input: [h_src(256) | h_dst(256) | hb+geom+pad(288)]
struct MsgAL {
  const _Float16* hl; const _Float16* rest; const int* sg; const int* dg;
  DEVFN h8 load8(int row, int k) const {
    if (k < H)     return *(const h8*)(hl + (size_t)sg[row] * H + k);
    if (k < 2 * H) return *(const h8*)(hl + (size_t)dg[row] * H + (k - H));
    return *(const h8*)(rest + (size_t)row * RW + (k - 2 * H));
  }
};
// cross input: [h_new broadcast over KC (256) | hPk+geom+pad(288)]
struct CrossAL {
  const _Float16* hn; const _Float16* rest;
  DEVFN h8 load8(int row, int k) const {
    if (k < H) return *(const h8*)(hn + (size_t)(row >> 5) * H + k);
    return *(const h8*)(rest + (size_t)row * RW + (k - H));
  }
};

// ---------------- generic WMMA GEMM: wave computes 32 rows x 64 cols --------------------
// grid = (M/256, N/64), block = 256 (8 waves). B strip [64 x Kp] staged in LDS.
// Each wave: 2 M-tiles x 4 N-tiles; B fragments loaded once per K-step, reused by 2 WMMAs.
template <class AL, int EPI>
__global__ __launch_bounds__(256) void gemm_wmma(
    AL al, const _Float16* __restrict__ Bt, const float* __restrict__ bias,
    float* __restrict__ outf, _Float16* __restrict__ outh,
    const int* __restrict__ idx, const float* __restrict__ scale,
    int N, int Kp)
{
  extern __shared__ _Float16 Bs[];
  const int n0 = blockIdx.y * 64;
  const int chunks = Kp >> 3;                    // 8-half chunks per B row
  for (int i = threadIdx.x; i < 64 * chunks; i += 256) {
    int n = i / chunks, c = (i - n * chunks) << 3;
    *(h8*)(Bs + (size_t)n * Kp + c) = *(const h8*)(Bt + (size_t)(n0 + n) * Kp + c);
  }
  __syncthreads();

  const int wid  = threadIdx.x >> 5;
  const int lane = threadIdx.x & 31;
  const int lr   = lane & 15;
  const int hi   = lane >> 4;
  const int m0   = (blockIdx.x * 8 + wid) * 32;
  const int arow = m0 + lr;

  v8f_t acc[2][4] = {};
  for (int k0 = 0; k0 < Kp; k0 += 32) {
    // ---- load phase: 2 A fragments + 4 B fragments into registers ----
    // A 16x32 f16 fragment per ISA layout: lanes0-15 K{k0..k0+7, k0+16..23}; lanes16-31 +8
    v16h_t af0 = mk16(al.load8(arow,      k0 + hi * 8),
                      al.load8(arow,      k0 + 16 + hi * 8));
    v16h_t af1 = mk16(al.load8(arow + 16, k0 + hi * 8),
                      al.load8(arow + 16, k0 + 16 + hi * 8));
    v16h_t bf[4];
#pragma unroll
    for (int j = 0; j < 4; ++j) {
      // B 32x16 f16 fragment: lanes0-15 K{k0..k0+7, k0+16..23}; lanes16-31 +8 (col = lr)
      const _Float16* bp = Bs + (size_t)(j * 16 + lr) * Kp + k0 + hi * 8;
      bf[j] = mk16(*(const h8*)bp, *(const h8*)(bp + 16));
    }
    // ---- compute phase: 8 back-to-back WMMAs ----
#pragma unroll
    for (int j = 0; j < 4; ++j) {
      acc[0][j] = __builtin_amdgcn_wmma_f32_16x16x32_f16(false, af0, false, bf[j],
                                                         (short)0, acc[0][j], false, false);
      acc[1][j] = __builtin_amdgcn_wmma_f32_16x16x32_f16(false, af1, false, bf[j],
                                                         (short)0, acc[1][j], false, false);
    }
  }

#pragma unroll
  for (int mt = 0; mt < 2; ++mt) {
#pragma unroll
    for (int j = 0; j < 4; ++j) {
      const int col = n0 + j * 16 + lr;
      const float bv = bias ? bias[col] : 0.f;
#pragma unroll
      for (int r = 0; r < 8; ++r) {
        const int row = m0 + mt * 16 + hi * 8 + r;  // C layout: VGPR r -> M=r / r+8
        float v = acc[mt][j][r] + bv;
        if (EPI == EPI_F16_RELU)     { v = v > 0.f ? v : 0.f; outh[(size_t)row * N + col] = (_Float16)v; }
        else if (EPI == EPI_F16)     { outh[(size_t)row * N + col] = (_Float16)v; }
        else if (EPI == EPI_F32)     { outf[(size_t)row * N + col] = v; }
        else if (EPI == EPI_SCATTER) { atomicAdd(outf + (size_t)idx[row] * N + col, v * scale[row]); }
        else                         { atomicAdd(outf + (size_t)(row >> 5) * N + col, v * scale[row]); }
      }
    }
  }
}

// ---------------- small utility kernels -------------------------------------------------
__global__ void prep_wt(const float* __restrict__ src, _Float16* __restrict__ dst,
                        int K, int N, int Kp) {
  int i = blockIdx.x * blockDim.x + threadIdx.x;
  if (i >= N * Kp) return;
  int n = i / Kp, k = i - n * Kp;
  dst[i] = (k < K) ? (_Float16)src[(size_t)k * N + n] : (_Float16)0.f;
}

__global__ void f32_to_f16_k(const float* __restrict__ s, _Float16* __restrict__ d, int n) {
  int i = blockIdx.x * blockDim.x + threadIdx.x;
  if (i < n) d[i] = (_Float16)s[i];
}

__global__ void zero_f32(float* __restrict__ p, int n) {
  int i = blockIdx.x * blockDim.x + threadIdx.x;
  if (i < n) p[i] = 0.f;
}

__global__ void time_embed(const float* __restrict__ t, const float* __restrict__ tW,
                           const float* __restrict__ tb, float* __restrict__ ht) {
  int b = blockIdx.x, h = threadIdx.x;
  float tmax = 1.f;
  for (int i = 0; i < Bn; ++i) tmax = fmaxf(tmax, t[i]);
  float tt = t[b] / tmax;
  float acc = tb[h];
  const float lg = logf(10000.f) / 127.f;
  for (int j = 0; j < 128; ++j) {
    float a = tt * expf(-j * lg);
    acc += sinf(a) * tW[(size_t)j * H + h] + cosf(a) * tW[(size_t)(j + 128) * H + h];
  }
  ht[(size_t)b * H + h] = acc;
}

__global__ void init_hL(const float* __restrict__ Xt, const int* __restrict__ At,
                        const float* __restrict__ aemb, const float* __restrict__ cW,
                        const float* __restrict__ cb, const float* __restrict__ ht,
                        float* __restrict__ hLf, _Float16* __restrict__ hLh) {
  int n = blockIdx.x, h = threadIdx.x, b = n / Nl;
  int a = At[n]; a = a < 0 ? 0 : (a > KA - 1 ? KA - 1 : a);
  float v = aemb[(size_t)a * H + h]
          + Xt[n * 3 + 0] * cW[h] + Xt[n * 3 + 1] * cW[H + h] + Xt[n * 3 + 2] * cW[2 * H + h]
          + cb[h] + ht[(size_t)b * H + h];
  hLf[(size_t)n * H + h] = v;
  hLh[(size_t)n * H + h] = (_Float16)v;
}

__global__ void knn_kernel(const float* __restrict__ Xt, const float* __restrict__ Xp,
                           const float* __restrict__ pmask, int* __restrict__ knn) {
  int id = blockIdx.x * blockDim.x + threadIdx.x;
  if (id >= NLIG) return;
  int b = id / Nl;
  float x0 = Xt[id * 3], x1 = Xt[id * 3 + 1], x2 = Xt[id * 3 + 2];
  float best[KC]; int bi[KC];
  for (int i = 0; i < KC; ++i) { best[i] = 3.0e38f; bi[i] = b * NP; }
  for (int p = 0; p < NP; ++p) {
    const float* xp = Xp + ((size_t)b * NP + p) * 3;
    float dx = xp[0] - x0, dy = xp[1] - x1, dz = xp[2] - x2;
    float d = sqrtf(dx * dx + dy * dy + dz * dz) + (1.f - pmask[(size_t)b * NP + p]) * 1e6f;
    if (d < best[KC - 1]) {
      int j = KC - 1;
      while (j > 0 && best[j - 1] > d) { best[j] = best[j - 1]; bi[j] = bi[j - 1]; --j; }
      best[j] = d; bi[j] = b * NP + p;
    }
  }
  for (int k = 0; k < KC; ++k) knn[(size_t)id * KC + k] = bi[k];
}

__global__ void build_cross(const float* __restrict__ Xt, const float* __restrict__ Xp,
                            const _Float16* __restrict__ hP, const int* __restrict__ knn,
                            _Float16* __restrict__ rest, float* __restrict__ w) {
  int row = blockIdx.x, t = threadIdx.x;
  int node = row / KC;
  int p = knn[row];
  if (t < H) { rest[(size_t)row * RW + t] = hP[(size_t)p * H + t]; return; }
  float rx = Xp[p * 3 + 0] - Xt[node * 3 + 0];
  float ry = Xp[p * 3 + 1] - Xt[node * 3 + 1];
  float rz = Xp[p * 3 + 2] - Xt[node * 3 + 2];
  float dist = fmaxf(sqrtf(rx * rx + ry * ry + rz * rz), 1e-6f);
  int c = t - H;
  _Float16 v = (_Float16)0.f;
  if (c == 0) { v = (_Float16)dist; w[row] = fminf(1.f / dist, 10.f); }
  else if (c == 1) v = (_Float16)(rx / dist);
  else if (c == 2) v = (_Float16)(ry / dist);
  else if (c == 3) v = (_Float16)(rz / dist);
  rest[(size_t)row * RW + t] = v;
}

__global__ void build_msg(const float* __restrict__ Xt, const int* __restrict__ bsrc,
                          const int* __restrict__ bdst, const int* __restrict__ btv,
                          const float* __restrict__ bemb, _Float16* __restrict__ rest,
                          int* __restrict__ sg, int* __restrict__ dg) {
  int e = blockIdx.x, t = threadIdx.x;
  int b = e / Ee;
  int sn = b * Nl + bsrc[e], dn = b * Nl + bdst[e];
  if (t == 0) { sg[e] = sn; dg[e] = dn; }
  if (t < H) {
    int bt = btv[e]; bt = bt < 0 ? 0 : (bt > KB - 1 ? KB - 1 : bt);
    rest[(size_t)e * RW + t] = (_Float16)bemb[(size_t)bt * H + t];
    return;
  }
  float rx = Xt[dn * 3 + 0] - Xt[sn * 3 + 0];
  float ry = Xt[dn * 3 + 1] - Xt[sn * 3 + 1];
  float rz = Xt[dn * 3 + 2] - Xt[sn * 3 + 2];
  float dist = fmaxf(sqrtf(rx * rx + ry * ry + rz * rz), 1e-6f);
  int c = t - H;
  _Float16 v = (_Float16)0.f;
  if (c == 0) v = (_Float16)dist;
  else if (c == 1) v = (_Float16)(rx / dist);
  else if (c == 2) v = (_Float16)(ry / dist);
  else if (c == 3) v = (_Float16)(rz / dist);
  rest[(size_t)e * RW + t] = v;
}

__global__ void gru_gate(const float* __restrict__ gx, const float* __restrict__ gh,
                         const float* __restrict__ hl, float* __restrict__ hnf,
                         _Float16* __restrict__ hnh) {
  int n = blockIdx.x, h = threadIdx.x;
  const float* gxp = gx + (size_t)n * G3H;
  const float* ghp = gh + (size_t)n * G3H;
  float r  = 1.f / (1.f + expf(-(gxp[h] + ghp[h])));
  float z  = 1.f / (1.f + expf(-(gxp[H + h] + ghp[H + h])));
  float ng = tanhf(gxp[2 * H + h] + r * ghp[2 * H + h]);
  float v  = (1.f - z) * ng + z * hl[(size_t)n * H + h];
  hnf[(size_t)n * H + h] = v;
  hnh[(size_t)n * H + h] = (_Float16)v;
}

__global__ void finalize_hL(const float* __restrict__ hn, const float* __restrict__ cacc,
                            const float* __restrict__ mask, float* __restrict__ hlf,
                            _Float16* __restrict__ hlh) {
  int n = blockIdx.x, h = threadIdx.x;
  float v = (hn[(size_t)n * H + h] + cacc[(size_t)n * H + h]) * mask[n];
  hlf[(size_t)n * H + h] = v;
  hlh[(size_t)n * H + h] = (_Float16)v;
}

__global__ void head_small(const _Float16* __restrict__ hin, const float* __restrict__ W,
                           const float* __restrict__ bvec, float* __restrict__ out,
                           int M, int C) {
  int id = blockIdx.x * blockDim.x + threadIdx.x;
  if (id >= M * C) return;
  int n = id / C, c = id - n * C;
  float acc = bvec[c];
  const _Float16* hp = hin + (size_t)n * H;
  for (int j = 0; j < H; ++j) acc += (float)hp[j] * W[(size_t)j * C + c];
  out[id] = acc;
}

// ---------------- host orchestration ----------------------------------------------------
extern "C" void kernel_launch(void* const* d_in, const int* in_sizes, int n_in,
                              void* d_out, int out_size, void* d_ws, size_t ws_size,
                              hipStream_t stream) {
  (void)in_sizes; (void)n_in; (void)out_size; (void)ws_size;
  const float* X_t  = (const float*)d_in[0];
  const int*   A_t  = (const int*)d_in[1];
  const int*   bsrc = (const int*)d_in[2];
  const int*   bdst = (const int*)d_in[3];
  const int*   B_tv = (const int*)d_in[4];
  const float* Xp   = (const float*)d_in[5];
  const float* Hp   = (const float*)d_in[6];
  const float* ligm = (const float*)d_in[7];
  const float* pokm = (const float*)d_in[8];
  const float* edgm = (const float*)d_in[9];
  const float* tvec = (const float*)d_in[10];
  const float* atom_emb = (const float*)d_in[11];
  const float* bond_emb = (const float*)d_in[12];
  const float* pocket_W = (const float*)d_in[13];
  const float* pocket_b = (const float*)d_in[14];
  const float* coord_W  = (const float*)d_in[15];
  const float* coord_b  = (const float*)d_in[16];
  const float* t_W = (const float*)d_in[17];
  const float* t_b = (const float*)d_in[18];
  const float* gru_Wih = (const float*)d_in[19];
  const float* gru_Whh = (const float*)d_in[20];
  const float* gru_bih = (const float*)d_in[21];
  const float* gru_bhh = (const float*)d_in[22];
  const float* msg_W1 = (const float*)d_in[23];
  const float* msg_b1 = (const float*)d_in[24];
  const float* msg_W2 = (const float*)d_in[25];
  const float* msg_b2 = (const float*)d_in[26];
  const float* cross_W1 = (const float*)d_in[27];
  const float* cross_b1 = (const float*)d_in[28];
  const float* cross_W2 = (const float*)d_in[29];
  const float* cross_b2 = (const float*)d_in[30];
  const float* eps_W1 = (const float*)d_in[31];
  const float* eps_b1 = (const float*)d_in[32];
  const float* eps_W2 = (const float*)d_in[33];
  const float* eps_b2 = (const float*)d_in[34];
  const float* A_W1 = (const float*)d_in[35];
  const float* A_b1 = (const float*)d_in[36];
  const float* A_W2 = (const float*)d_in[37];
  const float* A_b2 = (const float*)d_in[38];
  const float* B_W1 = (const float*)d_in[39];
  const float* B_b1 = (const float*)d_in[40];
  const float* B_W2 = (const float*)d_in[41];
  const float* B_b2 = (const float*)d_in[42];
  float* out = (float*)d_out;

  char* ws = (char*)d_ws; size_t off = 0;
  auto alloc = [&](size_t bytes) -> void* {
    void* p = ws + off; off += (bytes + 255) & ~(size_t)255; return p;
  };

  float*    ht     = (float*)alloc((size_t)Bn * H * 4);
  float*    hLf    = (float*)alloc((size_t)NLIG * H * 4);
  _Float16* hLh    = (_Float16*)alloc((size_t)NLIG * H * 2);
  _Float16* Hph    = (_Float16*)alloc((size_t)NPK * DP * 2);
  _Float16* hPh    = (_Float16*)alloc((size_t)NPK * H * 2);
  int*      knn    = (int*)alloc((size_t)NLIG * KC * 4);
  float*    wgt    = (float*)alloc((size_t)NCR * 4);
  _Float16* crest  = (_Float16*)alloc((size_t)NCR * RW * 2);
  _Float16* mrest  = (_Float16*)alloc((size_t)NE * RW * 2);
  int*      sg     = (int*)alloc((size_t)NE * 4);
  int*      dg     = (int*)alloc((size_t)NE * 4);
  float*    agg    = (float*)alloc((size_t)NLIG * H * 4);
  _Float16* aggh   = (_Float16*)alloc((size_t)NLIG * H * 2);
  _Float16* msgh   = (_Float16*)alloc((size_t)NE * H * 2);
  float*    gx     = (float*)alloc((size_t)NLIG * G3H * 4);
  float*    ghb    = (float*)alloc((size_t)NLIG * G3H * 4);
  float*    hnf    = (float*)alloc((size_t)NLIG * H * 4);
  _Float16* hnh    = (_Float16*)alloc((size_t)NLIG * H * 2);
  _Float16* crossh = (_Float16*)alloc((size_t)NCR * H * 2);
  float*    cacc   = (float*)alloc((size_t)NLIG * H * 4);
  _Float16* epsh   = (_Float16*)alloc((size_t)NLIG * H * 2);
  _Float16* Ah     = (_Float16*)alloc((size_t)NLIG * H * 2);
  _Float16* Bh     = (_Float16*)alloc((size_t)NE * H * 2);
  _Float16* pocketWt = (_Float16*)alloc((size_t)H * DP * 2);
  _Float16* WihT   = (_Float16*)alloc((size_t)L * G3H * H * 2);
  _Float16* WhhT   = (_Float16*)alloc((size_t)L * G3H * H * 2);
  _Float16* mW1T   = (_Float16*)alloc((size_t)L * H * KMSG * 2);
  _Float16* mW2T   = (_Float16*)alloc((size_t)L * H * H * 2);
  _Float16* cW1T   = (_Float16*)alloc((size_t)L * H * KCRS * 2);
  _Float16* cW2T   = (_Float16*)alloc((size_t)L * H * H * 2);
  _Float16* eW1T   = (_Float16*)alloc((size_t)H * H * 2);
  _Float16* aW1T   = (_Float16*)alloc((size_t)H * H * 2);
  _Float16* bW1T   = (_Float16*)alloc((size_t)H * KMSG * 2);

  auto prep = [&](const float* src, _Float16* dst, int K, int N, int Kp) {
    int tot = N * Kp;
    prep_wt<<<(tot + 255) / 256, 256, 0, stream>>>(src, dst, K, N, Kp);
  };

  // ---- weight transpose + f16 convert ----
  prep(pocket_W, pocketWt, DP, H, DP);
  prep(eps_W1, eW1T, H, H, H);
  prep(A_W1, aW1T, H, H, H);
  prep(B_W1, bW1T, 3 * H + 4, H, KMSG);
  for (int l = 0; l < L; ++l) {
    prep(gru_Wih + (size_t)l * H * G3H, WihT + (size_t)l * G3H * H, H, G3H, H);
    prep(gru_Whh + (size_t)l * H * G3H, WhhT + (size_t)l * G3H * H, H, G3H, H);
    prep(msg_W1 + (size_t)l * (3 * H + 4) * H, mW1T + (size_t)l * H * KMSG, 3 * H + 4, H, KMSG);
    prep(msg_W2 + (size_t)l * H * H, mW2T + (size_t)l * H * H, H, H, H);
    prep(cross_W1 + (size_t)l * (2 * H + 4) * H, cW1T + (size_t)l * H * KCRS, 2 * H + 4, H, KCRS);
    prep(cross_W2 + (size_t)l * H * H, cW2T + (size_t)l * H * H, H, H, H);
  }

  // ---- precompute ----
  f32_to_f16_k<<<(NPK * DP + 255) / 256, 256, 0, stream>>>(Hp, Hph, NPK * DP);
  time_embed<<<Bn, H, 0, stream>>>(tvec, t_W, t_b, ht);
  init_hL<<<NLIG, H, 0, stream>>>(X_t, A_t, atom_emb, coord_W, coord_b, ht, hLf, hLh);
  {
    PlainAL al{Hph, DP};
    dim3 g(NPK / 256, H / 64);
    gemm_wmma<PlainAL, EPI_F16><<<g, 256, 64 * DP * 2, stream>>>(
        al, pocketWt, pocket_b, nullptr, hPh, nullptr, nullptr, H, DP);
  }
  knn_kernel<<<(NLIG + 63) / 64, 64, 0, stream>>>(X_t, Xp, pokm, knn);
  build_cross<<<NCR, RW, 0, stream>>>(X_t, Xp, hPh, knn, crest, wgt);
  build_msg<<<NE, RW, 0, stream>>>(X_t, bsrc, bdst, B_tv, bond_emb, mrest, sg, dg);

  // ---- layers ----
  for (int l = 0; l < L; ++l) {
    zero_f32<<<(NLIG * H + 255) / 256, 256, 0, stream>>>(agg, NLIG * H);
    { MsgAL al{hLh, mrest, sg, dg};
      dim3 g(NE / 256, H / 64);
      gemm_wmma<MsgAL, EPI_F16_RELU><<<g, 256, 64 * KMSG * 2, stream>>>(
          al, mW1T + (size_t)l * H * KMSG, msg_b1 + (size_t)l * H,
          nullptr, msgh, nullptr, nullptr, H, KMSG);
    }
    { PlainAL al{msgh, H};
      dim3 g(NE / 256, H / 64);
      gemm_wmma<PlainAL, EPI_SCATTER><<<g, 256, 64 * H * 2, stream>>>(
          al, mW2T + (size_t)l * H * H, msg_b2 + (size_t)l * H,
          agg, nullptr, dg, edgm, H, H);
    }
    f32_to_f16_k<<<(NLIG * H + 255) / 256, 256, 0, stream>>>(agg, aggh, NLIG * H);
    { PlainAL al{aggh, H};
      dim3 g(NLIG / 256, G3H / 64);
      gemm_wmma<PlainAL, EPI_F32><<<g, 256, 64 * H * 2, stream>>>(
          al, WihT + (size_t)l * G3H * H, gru_bih + (size_t)l * G3H,
          gx, nullptr, nullptr, nullptr, G3H, H);
    }
    { PlainAL al{hLh, H};
      dim3 g(NLIG / 256, G3H / 64);
      gemm_wmma<PlainAL, EPI_F32><<<g, 256, 64 * H * 2, stream>>>(
          al, WhhT + (size_t)l * G3H * H, gru_bhh + (size_t)l * G3H,
          ghb, nullptr, nullptr, nullptr, G3H, H);
    }
    gru_gate<<<NLIG, H, 0, stream>>>(gx, ghb, hLf, hnf, hnh);
    { CrossAL al{hnh, crest};
      dim3 g(NCR / 256, H / 64);
      gemm_wmma<CrossAL, EPI_F16_RELU><<<g, 256, 64 * KCRS * 2, stream>>>(
          al, cW1T + (size_t)l * H * KCRS, cross_b1 + (size_t)l * H,
          nullptr, crossh, nullptr, nullptr, H, KCRS);
    }
    zero_f32<<<(NLIG * H + 255) / 256, 256, 0, stream>>>(cacc, NLIG * H);
    { PlainAL al{crossh, H};
      dim3 g(NCR / 256, H / 64);
      gemm_wmma<PlainAL, EPI_DIV32><<<g, 256, 64 * H * 2, stream>>>(
          al, cW2T + (size_t)l * H * H, cross_b2 + (size_t)l * H,
          cacc, nullptr, nullptr, wgt, H, H);
    }
    finalize_hL<<<NLIG, H, 0, stream>>>(hnf, cacc, ligm, hLf, hLh);
  }

  // ---- heads ----
  { PlainAL al{hLh, H};
    dim3 g(NLIG / 256, H / 64);
    gemm_wmma<PlainAL, EPI_F16_RELU><<<g, 256, 64 * H * 2, stream>>>(
        al, eW1T, eps_b1, nullptr, epsh, nullptr, nullptr, H, H);
    gemm_wmma<PlainAL, EPI_F16_RELU><<<g, 256, 64 * H * 2, stream>>>(
        al, aW1T, A_b1, nullptr, Ah, nullptr, nullptr, H, H);
  }
  { MsgAL al{hLh, mrest, sg, dg};
    dim3 g(NE / 256, H / 64);
    gemm_wmma<MsgAL, EPI_F16_RELU><<<g, 256, 64 * KMSG * 2, stream>>>(
        al, bW1T, B_b1, nullptr, Bh, nullptr, nullptr, H, KMSG);
  }
  head_small<<<(NLIG * 3 + 255) / 256, 256, 0, stream>>>(epsh, eps_W2, eps_b2, out, NLIG, 3);
  head_small<<<(NLIG * KA + 255) / 256, 256, 0, stream>>>(Ah, A_W2, A_b2, out + NLIG * 3, NLIG, KA);
  head_small<<<(NE * KB + 255) / 256, 256, 0, stream>>>(Bh, B_W2, B_b2,
                                                        out + NLIG * 3 + NLIG * KA, NE, KB);
}
